// Graph_14843406975266
// MI455X (gfx1250) — matrix operand order repaired and checked by reading, not compiled, not run on hardware
//
#include <hip/hip_runtime.h>
#include <hip/hip_bf16.h>
#include <math.h>

// Problem constants (from reference)
#define BB    256
#define TT    128
#define LBUF  128
#define LSTK  128
#define LHIST 256
#define WDIM  300
#define POSD  50
#define RELD  50
#define ACTD  50
#define FCD   128
#define UNIT  256
#define NACT  82

typedef __attribute__((ext_vector_type(16))) _Float16 v16h;
typedef __attribute__((ext_vector_type(8)))  _Float16 v8h;
typedef __attribute__((ext_vector_type(8)))  float    v8f;

__device__ __forceinline__ float sigf(float x) { return 1.0f / (1.0f + __expf(-x)); }

// ---------------------------------------------------------------------------
// Weight swizzle: f32 KxN row-major -> f16 fragment layout [kc][nt][lane][16]
//   element (lane,h) of block (kc,nt) holds W[kc*32 + (h>>3)*16 + (lane>>4)*8 + (h&7)][nt*16 + (lane&15)]
//   (zero-padded in K so WMMA K-chunks of 32 are exact)
// ---------------------------------------------------------------------------
__global__ void swizzle_kernel(const float* __restrict__ src, _Float16* __restrict__ dst,
                               int Kact, int N, int KC) {
    int NT = N >> 4;
    long total = (long)KC * NT * 32 * 16;
    for (long idx = (long)blockIdx.x * blockDim.x + threadIdx.x; idx < total;
         idx += (long)gridDim.x * blockDim.x) {
        int h    = (int)(idx & 15);
        int lane = (int)((idx >> 4) & 31);
        long rest = idx >> 9;
        int nt = (int)(rest % NT);
        int kc = (int)(rest / NT);
        int k = kc * 32 + ((h >> 3) << 4) + ((lane >> 4) << 3) + (h & 7);
        int n = (nt << 4) + (lane & 15);
        float v = (k < Kact) ? src[(long)k * N + n] : 0.0f;
        dst[idx] = (_Float16)v;
    }
}

// ---------------------------------------------------------------------------
// Wave-level GEMM: 16 rows (A in LDS, f16, row stride = strideH halves) times
// pre-swizzled weights (global f16), N = NT*16 columns, K = KC*32.
// 256-thread block = 8 waves; wave w computes ntiles [w*TPW, (w+1)*TPW).
// Result written as f16 to zout (row stride NT*16 halves).
// ---------------------------------------------------------------------------
template<int KC, int NT>
__device__ __forceinline__ void wave_gemm(const _Float16* __restrict__ Alds, int strideH,
                                          const _Float16* __restrict__ Wsw,
                                          _Float16* __restrict__ zout) {
    const int lane = threadIdx.x & 31;
    const int wave = threadIdx.x >> 5;
    const int g    = lane >> 4;        // lane-half
    const int mrow = lane & 15;        // A row / C column index
    constexpr int TPW = NT / 8;        // tiles per wave (8 waves)
    #pragma unroll
    for (int j = 0; j < TPW; ++j) {
        const int nt = wave * TPW + j;
        v8f acc = {};
        #pragma unroll
        for (int kc = 0; kc < KC; ++kc) {
            // A fragment (16x32 f16): lane-half g holds K = g*8..g*8+7 (VGPR0-3)
            // and K = 16+g*8..16+g*8+7 (VGPR4-7)   [ISA 7.12.2]
            const _Float16* ap = Alds + mrow * strideH + kc * 32 + g * 8;
            v8h alo = *(const v8h*)(ap);
            v8h ahi = *(const v8h*)(ap + 16);
            v16h a;
            #pragma unroll
            for (int e = 0; e < 8; ++e) { a[e] = alo[e]; a[e + 8] = ahi[e]; }
            // B fragment (32x16 f16): pre-swizzled, 16 contiguous halves per lane
            const _Float16* bp = Wsw + (((long)kc * NT + nt) * 32 + lane) * 16;
            v8h blo = *(const v8h*)(bp);
            v8h bhi = *(const v8h*)(bp + 8);
            v16h b;
            #pragma unroll
            for (int e = 0; e < 8; ++e) { b[e] = blo[e]; b[e + 8] = bhi[e]; }
            acc = __builtin_amdgcn_wmma_f32_16x16x32_f16(
                false, a, false, b, (short)0, acc, false, false);
        }
        // C layout: VGPR r -> row (r + 8*g), column lane&15  [ISA 7.12.2]
        const int n0 = (nt << 4) + mrow;
        #pragma unroll
        for (int r = 0; r < 8; ++r) {
            zout[(r + (g << 3)) * (NT * 16) + n0] = (_Float16)acc[r];
        }
    }
}

// ---------------------------------------------------------------------------
// Embedding: out[r] = relu([W_word[wid[r]], W_pos[pid[r]]] @ emb_W + emb_b), f16
// 16 rows per 256-thread workgroup. K = 350 padded to 352 (11 chunks).
// ---------------------------------------------------------------------------
__global__ __launch_bounds__(256)
void embed_kernel(const int* __restrict__ word_ids, const int* __restrict__ pos_ids,
                  const float* __restrict__ W_word, const float* __restrict__ W_pos,
                  const _Float16* __restrict__ embWsw, const float* __restrict__ emb_b,
                  _Float16* __restrict__ out) {
    __shared__ alignas(16) _Float16 Ae[16][352];
    __shared__ alignas(16) _Float16 ze[16][128];
    const int tid = threadIdx.x;
    const int r0  = blockIdx.x * 16;
    for (int i = tid; i < 16 * 352; i += 256) {
        int m = i / 352, k = i % 352;
        int r = r0 + m;
        float v = 0.0f;
        if (k < WDIM)             v = W_word[(long)word_ids[r] * WDIM + k];
        else if (k < WDIM + POSD) v = W_pos[pos_ids[r] * POSD + (k - WDIM)];
        Ae[m][k] = (_Float16)v;
    }
    __syncthreads();
    wave_gemm<11, 8>(&Ae[0][0], 352, embWsw, &ze[0][0]);
    __syncthreads();
    for (int i = tid; i < 16 * FCD; i += 256) {
        int m = i >> 7, n = i & 127;
        float v = (float)ze[m][n] + emb_b[n];
        out[(long)(r0 + m) * FCD + n] = (_Float16)fmaxf(v, 0.0f);
    }
}

// ---------------------------------------------------------------------------
// Recursive tree composition scan. One WG owns 16 batch rows; iterates t=1..127.
// res[b][t] = leaf ? tree_emb[b][t] : tanh([res[head],rel,res[dep]] @ rec_W + b)
// K = 306 padded to 320 (10 chunks).
// ---------------------------------------------------------------------------
__global__ __launch_bounds__(256)
void tree_scan_kernel(const _Float16* __restrict__ treeF16, const int* __restrict__ head,
                      const int* __restrict__ dep, const int* __restrict__ relid,
                      const int* __restrict__ leaf, const float* __restrict__ W_rel,
                      const _Float16* __restrict__ recWsw, const float* __restrict__ rec_b,
                      _Float16* __restrict__ resF16) {
    __shared__ alignas(16) _Float16 As[16][320];
    __shared__ alignas(16) _Float16 zs[16][128];
    const int tid = threadIdx.x;
    const int b0  = blockIdx.x * 16;
    for (int i = tid; i < 16 * 320; i += 256) {       // zero K padding [306,320)
        int m = i / 320, k = i % 320;
        if (k >= 306) As[m][k] = (_Float16)0.0f;
    }
    for (int i = tid; i < 16 * FCD; i += 256) {       // res[.,0] = tree_emb[:,0]
        int m = i >> 7, k = i & 127;
        long b = b0 + m;
        resF16[(b * TT) * FCD + k] = treeF16[(b * TT) * FCD + k];
    }
    __threadfence();
    __syncthreads();
    for (int t = 1; t < TT; ++t) {
        for (int i = tid; i < 16 * 306; i += 256) {
            int m = i / 306, k = i % 306;
            long b = b0 + m;
            _Float16 v;
            if (k < FCD) {
                int h = head[b * TT + t];
                v = resF16[(b * TT + h) * FCD + k];
            } else if (k < FCD + RELD) {
                int r = relid[b * TT + t];
                v = (_Float16)W_rel[r * RELD + (k - FCD)];
            } else {
                int d = dep[b * TT + t];
                v = resF16[(b * TT + d) * FCD + (k - FCD - RELD)];
            }
            As[m][k] = v;
        }
        __syncthreads();
        wave_gemm<10, 8>(&As[0][0], 320, recWsw, &zs[0][0]);
        __syncthreads();
        for (int i = tid; i < 16 * FCD; i += 256) {
            int m = i >> 7, n = i & 127;
            long b = b0 + m;
            _Float16 node;
            if (leaf[b * TT + t] > 0) node = treeF16[(b * TT + t) * FCD + n];
            else node = (_Float16)tanhf((float)zs[m][n] + rec_b[n]);
            resF16[(b * TT + t) * FCD + n] = node;
        }
        __threadfence();
        __syncthreads();
    }
}

// ---------------------------------------------------------------------------
// 2-layer masked LSTM over a sequence; emits last h2 (f32).
// One WG owns 16 batch rows; 256 threads = 8 waves; N = 1024 gate outputs.
// MODE 0: x = res[stack_order[b,t], b] (f16 gather)
// MODE 1: x = buff_emb[b, t]           (f16 direct)
// MODE 2: x = W_act[hist_id[b,t]]      (f32 gather -> f16)
// K1 = DIN+256 padded to K1C*32. Cell states live in registers (thread owns
// column n = tid for all 16 rows). h-states live in LDS (next A-matrices).
// ---------------------------------------------------------------------------
template<int DIN, int K1C, int MODE>
__global__ __launch_bounds__(256)
void lstm2_kernel(const _Float16* __restrict__ xF16, const float* __restrict__ xF32,
                  const int* __restrict__ idx, const int* __restrict__ lengths,
                  const _Float16* __restrict__ W1sw, const float* __restrict__ b1,
                  const _Float16* __restrict__ W2sw, const float* __restrict__ b2,
                  float* __restrict__ hout, int S) {
    constexpr int K1 = K1C * 32;
    __shared__ alignas(16) _Float16 A1[16][K1];     // [x | h1]   layer-1 A
    __shared__ alignas(16) _Float16 A2[16][512];    // [h1 | h2]  layer-2 A
    __shared__ alignas(16) _Float16 zbuf[16][1024]; // gate pre-activations
    const int tid = threadIdx.x;
    const int b0  = blockIdx.x * 16;
    const int n   = tid;                            // this thread's gate column
    float c1r[16], c2r[16];
    #pragma unroll
    for (int m = 0; m < 16; ++m) { c1r[m] = 0.0f; c2r[m] = 0.0f; }
    for (int i = tid; i < 16 * K1;  i += 256) ((_Float16*)A1)[i] = (_Float16)0.0f;
    for (int i = tid; i < 16 * 512; i += 256) ((_Float16*)A2)[i] = (_Float16)0.0f;
    __syncthreads();

    for (int t = 0; t < S; ++t) {
        // ---- gather x_t into A1[:, 0:DIN) ----
        if (MODE == 0) {
            for (int i = tid; i < 16 * FCD; i += 256) {
                int m = i >> 7, k = i & 127;
                long b = b0 + m;
                int ord = idx[b * LSTK + t];
                A1[m][k] = xF16[(b * TT + ord) * FCD + k];
            }
        } else if (MODE == 1) {
            for (int i = tid; i < 16 * FCD; i += 256) {
                int m = i >> 7, k = i & 127;
                long b = b0 + m;
                A1[m][k] = xF16[(b * LBUF + t) * FCD + k];
            }
        } else {
            for (int i = tid; i < 16 * ACTD; i += 256) {
                int m = i / ACTD, k = i % ACTD;
                long b = b0 + m;
                int id = idx[b * LHIST + t];
                A1[m][k] = (_Float16)xF32[id * ACTD + k];
            }
        }
        __syncthreads();
        // ---- layer 1: z = [x, h1] @ W1 ----
        wave_gemm<K1C, 64>(&A1[0][0], K1, W1sw, &zbuf[0][0]);
        __syncthreads();
        #pragma unroll
        for (int m = 0; m < 16; ++m) {
            if (t < lengths[b0 + m]) {
                float zi = (float)zbuf[m][n]            + b1[n];
                float zj = (float)zbuf[m][UNIT + n]     + b1[UNIT + n];
                float zf = (float)zbuf[m][2 * UNIT + n] + b1[2 * UNIT + n];
                float zo = (float)zbuf[m][3 * UNIT + n] + b1[3 * UNIT + n];
                float nc = sigf(zf + 1.0f) * c1r[m] + sigf(zi) * tanhf(zj);
                float nh = sigf(zo) * tanhf(nc);
                c1r[m] = nc;
                A1[m][DIN + n] = (_Float16)nh;   // h1 for next step's layer 1
                A2[m][n]       = (_Float16)nh;   // h1 for this step's layer 2
            }
        }
        __syncthreads();
        // ---- layer 2: z = [h1, h2] @ W2 ----
        wave_gemm<16, 64>(&A2[0][0], 512, W2sw, &zbuf[0][0]);
        __syncthreads();
        #pragma unroll
        for (int m = 0; m < 16; ++m) {
            if (t < lengths[b0 + m]) {
                float zi = (float)zbuf[m][n]            + b2[n];
                float zj = (float)zbuf[m][UNIT + n]     + b2[UNIT + n];
                float zf = (float)zbuf[m][2 * UNIT + n] + b2[2 * UNIT + n];
                float zo = (float)zbuf[m][3 * UNIT + n] + b2[3 * UNIT + n];
                float nc = sigf(zf + 1.0f) * c2r[m] + sigf(zi) * tanhf(zj);
                float nh = sigf(zo) * tanhf(nc);
                c2r[m] = nc;
                A2[m][UNIT + n] = (_Float16)nh;  // h2
            }
        }
        __syncthreads();
    }
    #pragma unroll
    for (int m = 0; m < 16; ++m) {
        hout[(long)(b0 + m) * UNIT + n] = (float)A2[m][UNIT + n];
    }
}

// ---------------------------------------------------------------------------
// Final projection: out = [s_h, b_h, a_h] @ fW + fb   (256 x 82)
// ---------------------------------------------------------------------------
__global__ void final_fc_kernel(const float* __restrict__ sh, const float* __restrict__ bh,
                                const float* __restrict__ ah, const float* __restrict__ fW,
                                const float* __restrict__ fb, float* __restrict__ out) {
    int i = blockIdx.x * blockDim.x + threadIdx.x;
    if (i >= BB * NACT) return;
    int b = i / NACT, o = i % NACT;
    float acc = fb[o];
    for (int k = 0; k < UNIT; ++k) acc += sh[b * UNIT + k] * fW[k * NACT + o];
    for (int k = 0; k < UNIT; ++k) acc += bh[b * UNIT + k] * fW[(UNIT + k) * NACT + o];
    for (int k = 0; k < UNIT; ++k) acc += ah[b * UNIT + k] * fW[(2 * UNIT + k) * NACT + o];
    out[i] = acc;
}

// ---------------------------------------------------------------------------
extern "C" void kernel_launch(void* const* d_in, const int* in_sizes, int n_in,
                              void* d_out, int out_size, void* d_ws, size_t ws_size,
                              hipStream_t stream) {
    (void)in_sizes; (void)n_in; (void)out_size; (void)ws_size;
    const int*   tree_word   = (const int*)d_in[0];
    const int*   tree_pos    = (const int*)d_in[1];
    const int*   buff_word   = (const int*)d_in[2];
    const int*   buff_pos    = (const int*)d_in[3];
    const int*   rel_id      = (const int*)d_in[4];
    const int*   hist_id     = (const int*)d_in[5];
    const int*   head_ord    = (const int*)d_in[6];
    const int*   dep_ord     = (const int*)d_in[7];
    const int*   is_leaf     = (const int*)d_in[8];
    const int*   stack_order = (const int*)d_in[9];
    const int*   stack_len   = (const int*)d_in[10];
    const int*   buff_len    = (const int*)d_in[11];
    const int*   hist_len    = (const int*)d_in[12];
    const float* W_word      = (const float*)d_in[13];
    const float* W_pos       = (const float*)d_in[14];
    const float* W_rel       = (const float*)d_in[15];
    const float* W_act       = (const float*)d_in[16];
    const float* emb_W       = (const float*)d_in[17];
    const float* emb_b       = (const float*)d_in[18];
    const float* rec_W       = (const float*)d_in[19];
    const float* rec_b       = (const float*)d_in[20];
    const float* sW1 = (const float*)d_in[21]; const float* sb1 = (const float*)d_in[22];
    const float* sW2 = (const float*)d_in[23]; const float* sb2 = (const float*)d_in[24];
    const float* bW1 = (const float*)d_in[25]; const float* bb1 = (const float*)d_in[26];
    const float* bW2 = (const float*)d_in[27]; const float* bb2 = (const float*)d_in[28];
    const float* aW1 = (const float*)d_in[29]; const float* ab1 = (const float*)d_in[30];
    const float* aW2 = (const float*)d_in[31]; const float* ab2 = (const float*)d_in[32];
    const float* fW  = (const float*)d_in[33]; const float* fb  = (const float*)d_in[34];

    // ---- workspace layout (~32 MB) ----
    char* ws = (char*)d_ws;
    size_t off = 0;
    auto take = [&](size_t bytes) -> char* {
        off = (off + 255) & ~(size_t)255;
        char* p = ws + off;
        off += bytes;
        return p;
    };
    _Float16* embWsw = (_Float16*)take((size_t)11 * 8  * 512 * 2);
    _Float16* recWsw = (_Float16*)take((size_t)10 * 8  * 512 * 2);
    _Float16* sW1sw  = (_Float16*)take((size_t)12 * 64 * 512 * 2);
    _Float16* sW2sw  = (_Float16*)take((size_t)16 * 64 * 512 * 2);
    _Float16* bW1sw  = (_Float16*)take((size_t)12 * 64 * 512 * 2);
    _Float16* bW2sw  = (_Float16*)take((size_t)16 * 64 * 512 * 2);
    _Float16* aW1sw  = (_Float16*)take((size_t)10 * 64 * 512 * 2);
    _Float16* aW2sw  = (_Float16*)take((size_t)16 * 64 * 512 * 2);
    _Float16* treeF16 = (_Float16*)take((size_t)BB * TT   * FCD * 2);
    _Float16* buffF16 = (_Float16*)take((size_t)BB * LBUF * FCD * 2);
    _Float16* resF16  = (_Float16*)take((size_t)BB * TT   * FCD * 2);
    float* sh = (float*)take((size_t)BB * UNIT * 4);
    float* bh = (float*)take((size_t)BB * UNIT * 4);
    float* ah = (float*)take((size_t)BB * UNIT * 4);

    // ---- 1) swizzle weights into WMMA B-fragment layout (once per call) ----
    auto sw = [&](const float* src, _Float16* dst, int Kact, int N, int KC) {
        long total = (long)KC * (N / 16) * 512;
        int blocks = (int)((total + 255) / 256);
        swizzle_kernel<<<blocks, 256, 0, stream>>>(src, dst, Kact, N, KC);
    };
    sw(emb_W, embWsw, 350, 128, 11);
    sw(rec_W, recWsw, 306, 128, 10);
    sw(sW1, sW1sw, 384, 1024, 12);  sw(sW2, sW2sw, 512, 1024, 16);
    sw(bW1, bW1sw, 384, 1024, 12);  sw(bW2, bW2sw, 512, 1024, 16);
    sw(aW1, aW1sw, 306, 1024, 10);  sw(aW2, aW2sw, 512, 1024, 16);

    // ---- 2) embeddings (WMMA GEMM + relu), f16 outputs ----
    embed_kernel<<<(BB * TT) / 16, 256, 0, stream>>>(
        tree_word, tree_pos, W_word, W_pos, embWsw, emb_b, treeF16);
    embed_kernel<<<(BB * LBUF) / 16, 256, 0, stream>>>(
        buff_word, buff_pos, W_word, W_pos, embWsw, emb_b, buffF16);

    // ---- 3) sequential recursive tree composition ----
    tree_scan_kernel<<<BB / 16, 256, 0, stream>>>(
        treeF16, head_ord, dep_ord, rel_id, is_leaf, W_rel, recWsw, rec_b, resF16);

    // ---- 4) three masked 2-layer LSTMs ----
    lstm2_kernel<128, 12, 0><<<BB / 16, 256, 0, stream>>>(
        resF16, nullptr, stack_order, stack_len, sW1sw, sb1, sW2sw, sb2, sh, LSTK);
    lstm2_kernel<128, 12, 1><<<BB / 16, 256, 0, stream>>>(
        buffF16, nullptr, nullptr, buff_len, bW1sw, bb1, bW2sw, bb2, bh, LBUF);
    lstm2_kernel<50, 10, 2><<<BB / 16, 256, 0, stream>>>(
        nullptr, W_act, hist_id, hist_len, aW1sw, ab1, aW2sw, ab2, ah, LHIST);

    // ---- 5) final projection ----
    final_fc_kernel<<<(BB * NACT + 255) / 256, 256, 0, stream>>>(
        sh, bh, ah, fW, fb, (float*)d_out);
}
